// KAttentionalPropagation_89739046683494
// MI455X (gfx1250) — matrix operand order repaired and checked by reading, not compiled, not run on hardware
//
#include <hip/hip_runtime.h>

// ---------------------------------------------------------------------------
// AttentionalPropagation on MI455X (gfx1250, wave32, WMMA 16x16x32 f16).
// All matmuls run on v_wmma_f32_16x16x32_f16 with f16 operands / f32 accum.
// Activations kept transposed ([row][channel], f16) so WMMA B-operand loads
// are two contiguous float4 per lane; weights row-major match the A-operand.
// GEMM waves compute a 16x64 strip (one A fragment reused by 4 B tiles) to
// raise WMMA-per-byte density. Attention is flash-style (transposed score
// tiles, online softmax with one cross-half shuffle) so the 256MB score
// tensor never exists.
// ---------------------------------------------------------------------------

typedef __attribute__((ext_vector_type(16))) _Float16 v16h;
typedef __attribute__((ext_vector_type(8)))  float    v8f;

union Frag16 { v16h v; float4 f4[2]; _Float16 e[16]; };
union Frag8f { v8f  v; float  e[8]; };

constexpr int   Bb  = 4;
constexpr int   Dd  = 256;
constexpr int   Hh  = 4;
constexpr int   Nn  = 2048;
constexpr int   Mm  = 2048;
constexpr int   HDc = 64;
constexpr int   D2  = 512;
constexpr float BN_SCALE = 0.9995003746877732f;  // 1/sqrt(1 + 1e-3)
constexpr float QK_SCALE = 0.125f;               // 1/sqrt(64)

__device__ __forceinline__ v8f zero8() {
  v8f z = {0.f, 0.f, 0.f, 0.f, 0.f, 0.f, 0.f, 0.f};
  return z;
}

__device__ __forceinline__ v8f wmma_f16(v16h a, v16h b, v8f c) {
  return __builtin_amdgcn_wmma_f32_16x16x32_f16(false, a, false, b,
                                                (short)0, c, false, false);
}

// A operand (16x32 f16). rowk -> element (row = lane&15, k = 0); k contiguous.
// Layout: lanes0-15 hold K[0..15]; lanes16-31 hold K[16..31].
__device__ __forceinline__ v16h load_a_frag(const _Float16* rowk, int hf) {
  Frag16 a;
  a.f4[0] = *(const float4*)(rowk + hf * 16);
  a.f4[1] = *(const float4*)(rowk + hf * 16 + 8);
  return a.v;
}

// B operand (32x16 f16). colk -> element (k = 0, col = lane&15); k contiguous
// (column-major source == row of a transposed activation tensor).
// Layout: lanes0-15 hold K[0..15]; lanes16-31 hold K[16..31].
__device__ __forceinline__ v16h load_b_frag(const _Float16* colk, int hf) {
  Frag16 b;
  b.f4[0] = *(const float4*)(colk + hf * 16);
  b.f4[1] = *(const float4*)(colk + hf * 16 + 8);
  return b.v;
}

__device__ __forceinline__ void load8f(const float* p, float* dst) {
  float4 a = *(const float4*)(p);
  float4 b = *(const float4*)(p + 4);
  dst[0] = a.x; dst[1] = a.y; dst[2] = a.z; dst[3] = a.w;
  dst[4] = b.x; dst[5] = b.y; dst[6] = b.z; dst[7] = b.w;
}

// ------------------------------- elementwise -------------------------------

__global__ __launch_bounds__(256)
void f32_to_f16_kernel(const float* __restrict__ in, _Float16* __restrict__ out,
                       int n) {
  int i = blockIdx.x * 256 + threadIdx.x;
  if (i < n) out[i] = (_Float16)in[i];
}

// in: [B][C][NC] f32 -> out[(b*NC+n)*ldo + colofs + c] f16 (transpose+convert)
__global__ __launch_bounds__(256)
void transpose_to_f16_kernel(const float* __restrict__ in,
                             _Float16* __restrict__ out,
                             int C, int NC, int ldo, int colofs) {
  int i = blockIdx.x * 256 + threadIdx.x;
  int total = Bb * C * NC;
  if (i >= total) return;
  int n = i % NC;
  int t = i / NC;
  int c = t % C;
  int b = t / C;
  out[((size_t)b * NC + n) * ldo + colofs + c] = (_Float16)in[i];
}

// --------------------------------- GEMM ------------------------------------
// Y = W @ X (+bias, epilogue per MODE). X given transposed: XT[b][n][k].
// Each wave computes a 16(cout) x 64(n) strip: 4 column tiles share one
// A fragment per K step (4 WMMAs per A load).
// MODE 0: f16 store YT[b][n][c0..]  (colofs within row of width ldy)
// MODE 1: f16 head-regroup store  YT[b][n][(c%H)*HD + c/H]   (Q, K)
// MODE 2: f16 store V layout      YT[((b*H + c%H)*HD + c/H)*NC + n]
// MODE 3: BN (gamma,beta) + ReLU, f16 store like MODE 0
// MODE 4: f32 store YF[(b*Cout + c)*NC + n]                  (final output)
template <int MODE>
__global__ __launch_bounds__(32)
void gemm_tn_kernel(const _Float16* __restrict__ W,   // [Cout][Cin]
                    const _Float16* __restrict__ XT,  // [B][NC][ldx]
                    const float* __restrict__ bias,
                    const float* __restrict__ bn_g,
                    const float* __restrict__ bn_b,
                    _Float16* __restrict__ YT,
                    float* __restrict__ YF,
                    int Cout, int Cin, int NC, int ldx, int ldy, int colofs) {
  const int tilesC = Cout >> 4;
  const int tilesN = NC >> 6;            // 64 columns per wave
  int bid = blockIdx.x;
  const int b = bid / (tilesC * tilesN);
  int rem = bid % (tilesC * tilesN);
  const int c0 = (rem / tilesN) * 16;
  const int n0 = (rem % tilesN) * 64;
  const int lane = threadIdx.x & 31;
  const int l15 = lane & 15;
  const int hf = lane >> 4;

  const _Float16* Wrow = W + (size_t)(c0 + l15) * Cin;
  const _Float16* Xcol = XT + (size_t)b * NC * ldx + (size_t)(n0 + l15) * ldx;
  const size_t tstride = (size_t)16 * ldx;

  v8f acc[4];
#pragma unroll
  for (int j = 0; j < 4; ++j) acc[j] = zero8();

  for (int k = 0; k < Cin; k += 32) {
    v16h a = load_a_frag(Wrow + k, hf);
#pragma unroll
    for (int j = 0; j < 4; ++j) {
      v16h bb = load_b_frag(Xcol + (size_t)j * tstride + k, hf);
      acc[j] = wmma_f16(a, bb, acc[j]);
    }
  }

#pragma unroll
  for (int j = 0; j < 4; ++j) {
    Frag8f r;
    r.v = acc[j];
    const int n = n0 + j * 16 + l15;
#pragma unroll
    for (int i = 0; i < 8; ++i) {
      const int c = c0 + i + 8 * hf;  // C/D layout: vgpr i = row i (+8 hi half)
      float y = r.e[i] + bias[c];
      if constexpr (MODE == 3) {
        y = fmaxf(y * (bn_g[c] * BN_SCALE) + bn_b[c], 0.f);
      }
      if constexpr (MODE == 0 || MODE == 3) {
        YT[((size_t)b * NC + n) * ldy + colofs + c] = (_Float16)y;
      } else if constexpr (MODE == 1) {
        const int cc = (c % Hh) * HDc + (c / Hh);
        YT[((size_t)b * NC + n) * ldy + cc] = (_Float16)y;
      } else if constexpr (MODE == 2) {
        const int hh = c % Hh, dp = c / Hh;
        YT[(((size_t)b * Hh + hh) * HDc + dp) * (size_t)NC + n] = (_Float16)y;
      } else if constexpr (MODE == 4) {
        YF[((size_t)b * Cout + c) * NC + n] = y;
      }
    }
  }
}

// ------------------------------ flash attention ----------------------------
// One wave per (b, h, 16-row n tile). Scores computed transposed:
// S^T[m,n] = sum_d kT[m,d] * qT[n,d]  (two 16x16x32 WMMAs per 16-m subtile).
// Softmax over m: 8-reg reduce + one cross-half shuffle. P fed back to WMMA
// as B operand against V (A operand) to accumulate msg^T[d',n].
__global__ __launch_bounds__(32)
void attn_flash_kernel(const _Float16* __restrict__ qT,  // [B][N][h*64+d']
                       const _Float16* __restrict__ kT,  // [B][M][h*64+d']
                       const _Float16* __restrict__ vH,  // [B][H][HD][M]
                       const float* __restrict__ wgt,    // [B][M]
                       const float* __restrict__ mask,   // [B][N][M]
                       _Float16* __restrict__ msgT) {    // [B][N][D], c=d'*H+h
  const int tilesN = Nn / 16;
  int bid = blockIdx.x;
  const int b = bid / (Hh * tilesN);
  int rem = bid % (Hh * tilesN);
  const int h = rem / tilesN;
  const int n0 = (rem % tilesN) * 16;
  const int lane = threadIdx.x & 31;
  const int l15 = lane & 15;
  const int hf = lane >> 4;

  // Stationary Q as B operand (K-dim = d', 64 -> two fragments).
  const _Float16* qrow = qT + ((size_t)b * Nn + n0 + l15) * Dd + h * HDc;
  v16h bq0 = load_b_frag(qrow, hf);
  v16h bq1 = load_b_frag(qrow + 32, hf);

  const _Float16* kbase = kT + ((size_t)b * Mm + l15) * Dd + h * HDc;
  const _Float16* vbase = vH + (((size_t)b * Hh + h) * HDc + l15) * (size_t)Mm;
  const float* mrow = mask + ((size_t)b * Nn + n0 + l15) * Mm + 8 * hf;
  const float* wrow = wgt + (size_t)b * Mm + 8 * hf;

  float run_max = -1e30f;
  float run_sum = 0.f;
  v8f acc[4];
#pragma unroll
  for (int t = 0; t < 4; ++t) acc[t] = zero8();

  for (int m0 = 0; m0 < Mm; m0 += 32) {
    __builtin_prefetch(kbase + (size_t)(m0 + 32) * Dd, 0, 0);

    // --- score subtiles S^T (rows m0..m0+15 and m0+16..m0+31) ---
    Frag8f s0, s1;
    {
      const _Float16* krow = kbase + (size_t)m0 * Dd;
      v16h a0 = load_a_frag(krow, hf);
      v16h a1 = load_a_frag(krow + 32, hf);
      s0.v = wmma_f16(a1, bq1, wmma_f16(a0, bq0, zero8()));
      krow += (size_t)16 * Dd;
      a0 = load_a_frag(krow, hf);
      a1 = load_a_frag(krow + 32, hf);
      s1.v = wmma_f16(a1, bq1, wmma_f16(a0, bq0, zero8()));
    }

    // --- scale * key-weight * mask (coalesced float4 loads) ---
    float mk0[8], mk1[8], w0[8], w1[8];
    load8f(mrow + m0, mk0);
    load8f(mrow + m0 + 16, mk1);
    load8f(wrow + m0, w0);
    load8f(wrow + m0 + 16, w1);
#pragma unroll
    for (int i = 0; i < 8; ++i) {
      s0.e[i] *= QK_SCALE * w0[i] * mk0[i];
      s1.e[i] *= QK_SCALE * w1[i] * mk1[i];
    }

    // --- online softmax over the 32-m chunk ---
    float cm = s0.e[0];
#pragma unroll
    for (int i = 1; i < 8; ++i) cm = fmaxf(cm, s0.e[i]);
#pragma unroll
    for (int i = 0; i < 8; ++i) cm = fmaxf(cm, s1.e[i]);
    cm = fmaxf(cm, __shfl_xor(cm, 16));
    const float nm = fmaxf(run_max, cm);
    const float corr = __expf(run_max - nm);
    run_max = nm;

    float cs = 0.f;
#pragma unroll
    for (int i = 0; i < 8; ++i) {
      s0.e[i] = __expf(s0.e[i] - nm);
      s1.e[i] = __expf(s1.e[i] - nm);
      cs += s0.e[i] + s1.e[i];
    }
    cs += __shfl_xor(cs, 16);
    run_sum = run_sum * corr + cs;
#pragma unroll
    for (int t = 0; t < 4; ++t) acc[t] = acc[t] * corr;

    // --- rebuild exp(S^T) chunk as a 32x16 B operand ---
    // Lane half 0 needs tile0 entirely, half 1 needs tile1; rows {0..7} and
    // {8..15} of the needed tile live in opposite lane halves -> one xor-16
    // shuffle delivers the missing 8 rows.
    Frag8f snd, rmt, loc;
    snd.v = hf ? s0.v : s1.v;   // send what the partner half needs
#pragma unroll
    for (int i = 0; i < 8; ++i) rmt.e[i] = __shfl_xor(snd.e[i], 16);
    loc.v = hf ? s1.v : s0.v;   // local slice of the tile this half needs

    Frag16 bp;
#pragma unroll
    for (int e = 0; e < 8; ++e) {
      float lo = hf ? rmt.e[e] : loc.e[e];        // rows 0..7 of needed tile
      float hi = hf ? loc.e[e] : rmt.e[e];        // rows 8..15
      bp.e[e] = (_Float16)lo;
      bp.e[e + 8] = (_Float16)hi;
    }

    // --- accumulate msg^T[d',n] += V(d',m) * P(m,n), 4 d'-subtiles ---
#pragma unroll
    for (int t = 0; t < 4; ++t) {
      v16h va = load_a_frag(vbase + (size_t)(t * 16) * Mm + m0, hf);
      acc[t] = wmma_f16(va, bp.v, acc[t]);
    }
  }

  // --- normalize and scatter to interleaved-channel msg^T ---
  const float inv = 1.f / run_sum;
  _Float16* orow = msgT + ((size_t)b * Nn + n0 + l15) * Dd;
#pragma unroll
  for (int t = 0; t < 4; ++t) {
    Frag8f fr;
    fr.v = acc[t];
#pragma unroll
    for (int i = 0; i < 8; ++i) {
      const int dp = t * 16 + i + 8 * hf;
      orow[dp * Hh + h] = (_Float16)(fr.e[i] * inv);
    }
  }
}

// --------------------------------- launch ----------------------------------

extern "C" void kernel_launch(void* const* d_in, const int* in_sizes, int n_in,
                              void* d_out, int out_size, void* d_ws,
                              size_t ws_size, hipStream_t stream) {
  const float* x    = (const float*)d_in[0];
  const float* src  = (const float*)d_in[1];
  const float* wgt  = (const float*)d_in[2];
  const float* mask = (const float*)d_in[3];
  const float* Wq   = (const float*)d_in[4];
  const float* bq   = (const float*)d_in[5];
  const float* Wk   = (const float*)d_in[6];
  const float* bk   = (const float*)d_in[7];
  const float* Wv   = (const float*)d_in[8];
  const float* bv   = (const float*)d_in[9];
  const float* Wm   = (const float*)d_in[10];
  const float* bm   = (const float*)d_in[11];
  const float* W1   = (const float*)d_in[12];
  const float* b1   = (const float*)d_in[13];
  const float* g1   = (const float*)d_in[14];
  const float* be1  = (const float*)d_in[15];
  const float* W2   = (const float*)d_in[16];
  const float* b2   = (const float*)d_in[17];
  float* out = (float*)d_out;

  char* ws = (char*)d_ws;
  size_t off = 0;
  auto take = [&](size_t bytes) -> char* {
    char* p = ws + off;
    off = (off + bytes + 255) & ~(size_t)255;
    return p;
  };
  _Float16* Wqh  = (_Float16*)take((size_t)Dd * Dd * 2);
  _Float16* Wkh  = (_Float16*)take((size_t)Dd * Dd * 2);
  _Float16* Wvh  = (_Float16*)take((size_t)Dd * Dd * 2);
  _Float16* Wmh  = (_Float16*)take((size_t)Dd * Dd * 2);
  _Float16* W1h  = (_Float16*)take((size_t)D2 * D2 * 2);
  _Float16* W2h  = (_Float16*)take((size_t)Dd * D2 * 2);
  _Float16* sT   = (_Float16*)take((size_t)Bb * Mm * Dd * 2);
  _Float16* catT = (_Float16*)take((size_t)Bb * Nn * D2 * 2);  // [x | merged]
  _Float16* qT   = (_Float16*)take((size_t)Bb * Nn * Dd * 2);
  _Float16* kTb  = (_Float16*)take((size_t)Bb * Mm * Dd * 2);
  _Float16* vHb  = (_Float16*)take((size_t)Bb * Mm * Dd * 2);
  _Float16* msgT = (_Float16*)take((size_t)Bb * Nn * Dd * 2);
  _Float16* h1T  = (_Float16*)take((size_t)Bb * Nn * D2 * 2);
  if (off > ws_size) return;  // workspace too small: bail deterministically

  const dim3 blk1(256), blkW(32);

  // fp32 -> f16 weight conversions
  f32_to_f16_kernel<<<(Dd * Dd + 255) / 256, blk1, 0, stream>>>(Wq, Wqh, Dd * Dd);
  f32_to_f16_kernel<<<(Dd * Dd + 255) / 256, blk1, 0, stream>>>(Wk, Wkh, Dd * Dd);
  f32_to_f16_kernel<<<(Dd * Dd + 255) / 256, blk1, 0, stream>>>(Wv, Wvh, Dd * Dd);
  f32_to_f16_kernel<<<(Dd * Dd + 255) / 256, blk1, 0, stream>>>(Wm, Wmh, Dd * Dd);
  f32_to_f16_kernel<<<(D2 * D2 + 255) / 256, blk1, 0, stream>>>(W1, W1h, D2 * D2);
  f32_to_f16_kernel<<<(Dd * D2 + 255) / 256, blk1, 0, stream>>>(W2, W2h, Dd * D2);

  // activations -> transposed f16 (x lands in left half of the concat buffer)
  transpose_to_f16_kernel<<<(Bb * Dd * Nn + 255) / 256, blk1, 0, stream>>>(
      x, catT, Dd, Nn, D2, 0);
  transpose_to_f16_kernel<<<(Bb * Dd * Mm + 255) / 256, blk1, 0, stream>>>(
      src, sT, Dd, Mm, Dd, 0);

  // Q/K/V projections (WMMA GEMMs, 16x64 strip per wave)
  const int gQ = Bb * (Dd / 16) * (Nn / 64);
  const int gK = Bb * (Dd / 16) * (Mm / 64);
  gemm_tn_kernel<1><<<gQ, blkW, 0, stream>>>(Wqh, catT, bq, nullptr, nullptr,
                                             qT, nullptr, Dd, Dd, Nn, D2, Dd, 0);
  gemm_tn_kernel<1><<<gK, blkW, 0, stream>>>(Wkh, sT, bk, nullptr, nullptr,
                                             kTb, nullptr, Dd, Dd, Mm, Dd, Dd, 0);
  gemm_tn_kernel<2><<<gK, blkW, 0, stream>>>(Wvh, sT, bv, nullptr, nullptr,
                                             vHb, nullptr, Dd, Dd, Mm, Dd, 0, 0);

  // flash attention
  attn_flash_kernel<<<Bb * Hh * (Nn / 16), blkW, 0, stream>>>(qT, kTb, vHb,
                                                              wgt, mask, msgT);

  // merge projection -> right half of the concat buffer
  gemm_tn_kernel<0><<<gQ, blkW, 0, stream>>>(Wmh, msgT, bm, nullptr, nullptr,
                                             catT, nullptr, Dd, Dd, Nn, Dd, D2,
                                             Dd);
  // MLP layer 1 (+BN eval +ReLU)
  const int gM1 = Bb * (D2 / 16) * (Nn / 64);
  gemm_tn_kernel<3><<<gM1, blkW, 0, stream>>>(W1h, catT, b1, g1, be1, h1T,
                                              nullptr, D2, D2, Nn, D2, D2, 0);
  // MLP layer 2 -> fp32 output [B][D][N]
  gemm_tn_kernel<4><<<gQ, blkW, 0, stream>>>(W2h, h1T, b2, nullptr, nullptr,
                                             nullptr, out, Dd, D2, Nn, D2, 0, 0);
}
